// ModulatedConv2d_89988154786113
// MI455X (gfx1250) — matrix operand order repaired and checked by reading, not compiled, not run on hardware
//
#include <hip/hip_runtime.h>
#include <hip/hip_bf16.h>
#include <stdint.h>

typedef __attribute__((ext_vector_type(8)))  _Float16 v8h;
typedef __attribute__((ext_vector_type(16))) _Float16 v16h;
typedef __attribute__((ext_vector_type(8)))  float    v8f;
typedef unsigned int su4 __attribute__((ext_vector_type(4)));
typedef int          si8 __attribute__((ext_vector_type(8)));
typedef int          si4 __attribute__((ext_vector_type(4)));

#define BATCH 8
#define CIN   512
#define COUT  512
#define NHW   4096          // 64*64
#define KK    4608          // CIN*9
#define SDIM  512
#define EPSV  1e-8f

#define BLK_M 128
#define BLK_N 128
#define BLK_K 32
#define LDSP  40            // halves per LDS row (32 + 8 pad) -> 80B stride
#define NSTEPS (KK / BLK_K) // 144

// ---- TDM tiers: 2 = clang builtin, 1 = inline asm, 0 = host-pass plain copy
#if defined(__gfx1250__)
#  if __has_builtin(__builtin_amdgcn_tensor_load_to_lds)
#    define USE_TDM 2
#  else
#    define USE_TDM 1
#  endif
#else
#  define USE_TDM 0
#endif

#if USE_TDM == 2
#  if __has_include(<hip/amd_detail/amd_gfx1250_TDM.h>)
#    define TDM_LOAD(g0, g1, z4) \
       __builtin_amdgcn_tensor_load_to_lds((g0), (g1), (z4), (z4), (si8){0,0,0,0,0,0,0,0}, 0)
#    warning "CDNA5 probe: TDM via 6-arg builtin"
#  else
#    define TDM_LOAD(g0, g1, z4) \
       __builtin_amdgcn_tensor_load_to_lds((g0), (g1), (z4), (z4), 0)
#    warning "CDNA5 probe: TDM via 5-arg builtin"
#  endif
#elif USE_TDM == 1
#  warning "CDNA5 probe: TDM via inline asm (builtin unavailable)"
#endif

#if USE_TDM > 0
#  if __has_builtin(__builtin_amdgcn_s_wait_tensorcnt)
#    define TENSOR_WAIT() __builtin_amdgcn_s_wait_tensorcnt(0)
#  else
#    define TENSOR_WAIT() asm volatile("s_wait_tensorcnt 0x0" ::: "memory")
#  endif
#endif

// ---------------- Kernel 1: style modulation  s[b][ci] = style[b,:]·w_mod[ci,:] + b_mod[ci]
__global__ void style_mod_kernel(const float* __restrict__ style,
                                 const float* __restrict__ w_mod,
                                 const float* __restrict__ b_mod,
                                 float* __restrict__ s) {
    int idx = blockIdx.x * blockDim.x + threadIdx.x;
    if (idx >= BATCH * CIN) return;
    int b = idx / CIN, ci = idx - b * CIN;
    const float* st = style + (size_t)b * SDIM;
    const float* wr = w_mod + (size_t)ci * SDIM;
    float acc = b_mod[ci];
    for (int j = 0; j < SDIM; ++j) acc += st[j] * wr[j];
    s[idx] = acc;
}

// ---------------- Kernel 2: demodulate + pack f16 weights  wf16[b][co][k]
__global__ void demod_pack_kernel(const float* __restrict__ weight,
                                  const float* __restrict__ s,
                                  _Float16* __restrict__ wf16) {
    int co  = blockIdx.x;
    int b   = blockIdx.y;
    int tid = threadIdx.x;                                // 256 threads
    const float* wrow = weight + (size_t)co * KK;
    const float* sb   = s + (size_t)b * CIN;

    float vals[KK / 256];                                 // 18 per thread
    float ss = 0.f;
    int i = 0;
    for (int k = tid; k < KK; k += 256, ++i) {
        int ci = k / 9;
        float v = wrow[k] * sb[ci];
        vals[i] = v;
        ss += v * v;
    }
    __shared__ float red[256];
    red[tid] = ss;
    __syncthreads();
    for (int off = 128; off > 0; off >>= 1) {
        if (tid < off) red[tid] += red[tid + off];
        __syncthreads();
    }
    float demod = rsqrtf(red[0] + EPSV);
    _Float16* orow = wf16 + (size_t)(b * COUT + co) * KK;
    i = 0;
    for (int k = tid; k < KK; k += 256, ++i)
        orow[k] = (_Float16)(vals[i] * demod);
}

// ---------------- Kernel 3: implicit-GEMM conv, WMMA f16 + TDM A-tiles, double-buffered LDS
__global__ __launch_bounds__(256)
void modconv_wmma_kernel(const float* __restrict__ x,
                         const _Float16* __restrict__ wf16,
                         float* __restrict__ out) {
    __shared__ __align__(16) _Float16 As[2][BLK_M * LDSP];   // weights tile [m][k]
    __shared__ __align__(16) _Float16 Bt[2][BLK_N * LDSP];   // im2col tile  [n][k]

    const int b    = blockIdx.z;
    const int m0   = blockIdx.y * BLK_M;
    const int n0   = blockIdx.x * BLK_N;
    const int tid  = threadIdx.x;
    const int wid  = tid >> 5;
    const int lane = tid & 31;
    const int wm   = (wid >> 1) * 32;    // wave tile 32(M) x 64(N); 4x2 waves
    const int wn   = (wid & 1)  * 64;

    v8f acc[2][4];
    v8f zero = {0.f, 0.f, 0.f, 0.f, 0.f, 0.f, 0.f, 0.f};
    for (int im = 0; im < 2; ++im)
        for (int in = 0; in < 4; ++in) acc[im][in] = zero;

    // ---- B staging: thread owns TWO adjacent k-slices x 8 consecutive n (same image row)
    const int kpair = (tid >> 4) * 2;      // 0,2,...,30
    const int nb    = (tid & 15) * 8;      // n_local base, 0..120
    const int hbase = (n0 + nb) >> 6;      // constant image row for the 8 n
    const int wbase = (n0 + nb) & 63;      // in {0,8,...,56}
    const float* xb = x + (size_t)b * CIN * NHW;

    // incremental k = step*32 + kpair -> (ci0, r0); 32 = 3*9 + 5
    int ci0 = kpair / 9;
    int r0  = kpair - ci0 * 9;

    const _Float16* a_tile = wf16 + (size_t)(b * COUT + m0) * KK;

#if USE_TDM == 0
    const int a_row  = tid >> 1;
    const int a_half = (tid & 1) * 16;
    const _Float16* a_g = a_tile + (size_t)a_row * KK + a_half;
#endif

    const int mrow  = lane & 15;
    const int khalf = lane >> 4;
    union V16 { v16h v; v8h h[2]; };

    auto bload1 = [&](int cij, int rj, float* breg) {   // 8 floats for one k-slice
        int q  = rj / 3;
        int dh = q - 1;
        int dw = (rj - q * 3) - 1;
        int hh = hbase + dh;
        int w0 = wbase + dw;
        const float* rp = xb + (size_t)cij * NHW + hh * 64;
        if ((unsigned)hh < 64u) {
            breg[0] = ((unsigned)w0 < 64u) ? rp[w0] : 0.f;
#pragma unroll
            for (int t = 1; t < 7; ++t) breg[t] = rp[w0 + t];
            breg[7] = ((unsigned)(w0 + 7) < 64u) ? rp[w0 + 7] : 0.f;
        } else {
#pragma unroll
            for (int t = 0; t < 8; ++t) breg[t] = 0.f;
        }
    };
    auto bload = [&](float* breg) {                     // 16 floats: k-slices kpair,kpair+1
        int r1 = r0 + 1, ci1 = ci0;
        if (r1 == 9) { r1 = 0; ci1 += 1; }
        bload1(ci0, r0, breg);
        bload1(ci1, r1, breg + 8);
    };
    auto bstore = [&](const float* breg, int pbuf) {    // 8 packed b32 stores
        _Float16* bp = &Bt[pbuf][nb * LDSP + kpair];
#pragma unroll
        for (int t = 0; t < 8; ++t) {
            union { _Float16 h[2]; uint32_t u; } pk;
            pk.h[0] = (_Float16)breg[t];
            pk.h[1] = (_Float16)breg[t + 8];
            *(uint32_t*)(bp + t * LDSP) = pk.u;         // kpair even -> 4B aligned
        }
    };
    auto kadv = [&]() { ci0 += 3; r0 += 5; if (r0 >= 9) { r0 -= 9; ci0 += 1; } };

#if USE_TDM > 0
    // TDM: load 128x32-half tile, row stride KK halves, LDS pad 4 DWORDs every 16 DWORDs
    auto tdm_issue = [&](int k0n, int pbuf) {
        uint64_t ga = (uint64_t)(uintptr_t)(a_tile + k0n);
        su4 g0;
        g0.x = 1u;                                                // count=1 valid
        g0.y = (uint32_t)(uintptr_t)(&As[pbuf][0]);               // lds_addr
        g0.z = (uint32_t)ga;                                      // global_addr[31:0]
        g0.w = ((uint32_t)(ga >> 32) & 0x01FFFFFFu) | (2u << 30); // addr[56:32] | type=2
        si8 g1 = {
            (int)0x06D10000,   // data_size=2B, pad_enable, pad_interval=16dw, pad_amount=4dw
            (int)0x12000000,   // tensor_dim0 = 4608 (low 16 bits in [31:16])
            (int)0x02000000,   // tensor_dim1 = 512 (low 16 bits in [31:16])
            (int)(32 << 16),   // tile_dim0 = 32 halves
            128,               // tile_dim1 = 128 rows
            4608,              // tensor_dim0_stride[31:0]
            0, 0
        };
#if USE_TDM == 2
        si4 z4 = {0, 0, 0, 0};
        TDM_LOAD(g0, g1, z4);
#else
        asm volatile("tensor_load_to_lds %0, %1" :: "s"(g0), "s"(g1) : "memory");
#endif
    };
#endif

    // ---- prologue: stage step 0 into buffer 0
    {
        float breg[16];
        bload(breg);
#if USE_TDM > 0
        if (wid == 0) tdm_issue(0, 0);
#else
        {
            const uint4* src = (const uint4*)(a_g);
            uint4* dst = (uint4*)(&As[0][a_row * LDSP + a_half]);
            dst[0] = src[0];
            dst[1] = src[1];
        }
#endif
        bstore(breg, 0);
        kadv();
#if USE_TDM > 0
        if (wid == 0) TENSOR_WAIT();
#endif
        __syncthreads();
    }

    // ---- main pipelined loop
    for (int i = 0; i < NSTEPS; ++i) {
        const int p = i & 1;
        const bool more = (i + 1) < NSTEPS;
        float breg[16];
        if (more) {
            bload(breg);                       // issue globals for step i+1 early
#if USE_TDM > 0
            if (wid == 0) tdm_issue((i + 1) * BLK_K, p ^ 1);
#else
            {
                const uint4* src = (const uint4*)(a_g + (i + 1) * BLK_K);
                uint4* dst = (uint4*)(&As[p ^ 1][a_row * LDSP + a_half]);
                dst[0] = src[0];
                dst[1] = src[1];
            }
#endif
        }

        // ---- compute step i from buffer p (ISA fragment layouts)
        V16 af[2], bf[4];
        for (int im = 0; im < 2; ++im) {
            const _Float16* pa = &As[p][(wm + im * 16 + mrow) * LDSP + khalf * 8];
            af[im].h[0] = *(const v8h*)pa;          // K = khalf*8 + 0..7
            af[im].h[1] = *(const v8h*)(pa + 16);   // K = 16 + khalf*8 + 0..7
        }
        for (int in = 0; in < 4; ++in) {
            const _Float16* pb = &Bt[p][(wn + in * 16 + mrow) * LDSP + khalf * 8];
            bf[in].h[0] = *(const v8h*)pb;          // K = khalf*8 + 0..7
            bf[in].h[1] = *(const v8h*)(pb + 16);   // K = 16 + khalf*8 + 0..7
        }
        for (int im = 0; im < 2; ++im)
            for (int in = 0; in < 4; ++in)
                acc[im][in] = __builtin_amdgcn_wmma_f32_16x16x32_f16(
                    false, af[im].v, false, bf[in].v,
                    (short)0, acc[im][in], false, false);

        if (more) {
            bstore(breg, p ^ 1);
            kadv();
#if USE_TDM > 0
            if (wid == 0) TENSOR_WAIT();
#endif
            __syncthreads();
        }
    }

    // ---- epilogue: C layout VGPR rr -> M = rr + (lane>>4)*8, N = lane&15
    const int mh = (lane >> 4) * 8;
    float* ob = out + (size_t)b * COUT * NHW;
    for (int im = 0; im < 2; ++im)
        for (int in = 0; in < 4; ++in) {
            int mbase = m0 + wm + im * 16 + mh;
            int nidx  = n0 + wn + in * 16 + mrow;
            for (int rr = 0; rr < 8; ++rr)
                ob[(size_t)(mbase + rr) * NHW + nidx] = acc[im][in][rr];
        }
}

extern "C" void kernel_launch(void* const* d_in, const int* in_sizes, int n_in,
                              void* d_out, int out_size, void* d_ws, size_t ws_size,
                              hipStream_t stream) {
    const float* x      = (const float*)d_in[0];
    const float* style  = (const float*)d_in[1];
    const float* w_mod  = (const float*)d_in[2];
    const float* b_mod  = (const float*)d_in[3];
    const float* weight = (const float*)d_in[4];
    float* out = (float*)d_out;

    float*    s    = (float*)d_ws;                                           // 16 KB
    _Float16* wf16 = (_Float16*)((char*)d_ws + BATCH * CIN * sizeof(float)); // 37.7 MB

    style_mod_kernel<<<(BATCH * CIN + 255) / 256, 256, 0, stream>>>(style, w_mod, b_mod, s);
    demod_pack_kernel<<<dim3(COUT, BATCH), 256, 0, stream>>>(weight, s, wf16);
    modconv_wmma_kernel<<<dim3(NHW / BLK_N, COUT / BLK_M, BATCH), 256, 0, stream>>>(x, wf16, out);
}
